// ReadMemory_15453292331556
// MI455X (gfx1250) — matrix operand order repaired and checked by reading, not compiled, not run on hardware
//
#include <hip/hip_runtime.h>
#include <cstdint>

#define B_SZ    128
#define DMODEL  1024
#define DMEM    64
#define NMEM    4096
#define CHUNK   128
#define PADC    144  // CHUNK+16 -> 288B row stride, multiple of 16B for b128 LDS ops
#define PADK    80

typedef __bf16 bf16_t;
typedef bf16_t v16bf __attribute__((ext_vector_type(16)));
typedef float  v8f   __attribute__((ext_vector_type(8)));

// ---- helpers -------------------------------------------------------------

// Pack two floats to bf16x2 (hardware cvt) and store as one 32-bit LDS word.
__device__ __forceinline__ void store_pk(bf16_t* dst, float a, float b) {
  union { bf16_t h[2]; uint32_t u; } pk;
  pk.h[0] = (bf16_t)a;
  pk.h[1] = (bf16_t)b;
  *(uint32_t*)dst = pk.u;
}

// A fragment: 16x32 bf16 tile, row-major, stride lda (elements).
// Lanes 0-15 row M=lane hold K=0-15; lanes 16-31 same rows hold K=16-31.
__device__ __forceinline__ v16bf load_frag_a(const bf16_t* a, int lda, int lane) {
  const int r = lane & 15;
  const int h = (lane >> 4) & 1;
  const bf16_t* p = a + r * lda + h * 16;
  v16bf out;
#pragma unroll
  for (int j = 0; j < 16; ++j) out[j] = p[j];
  return out;
}

// B fragment: 32x16 bf16 tile where B[k][n] = Wt[n][k], Wt row-major stride ldw.
// Lanes 0-15 col N=lane hold K=0-15 (2 per VGPR), lanes 16-31 hold K=16-31.
__device__ __forceinline__ v16bf load_frag_b(const bf16_t* wt, int ldw, int lane) {
  const int c = lane & 15;
  const int h = (lane >> 4) & 1;
  const bf16_t* p = wt + c * ldw + h * 16;
  v16bf out;
#pragma unroll
  for (int j = 0; j < 16; ++j) out[j] = p[j];
  return out;
}

__device__ __forceinline__ v8f wmma_bf16(v16bf a, v16bf b, v8f c) {
  return __builtin_amdgcn_wmma_f32_16x16x32_bf16(false, a, false, b, (short)0, c,
                                                 false, false);
}

// Async DMA one CHUNKxDMEM f32 tile (32KB) global -> LDS. 8 b128 ops per lane.
__device__ __forceinline__ void issue_chunk(const float* gbase, int c,
                                            float* ldsbuf, int tid) {
  uint32_t lds0 = (uint32_t)(uintptr_t)ldsbuf + (uint32_t)(tid * 16);
  uint32_t goff = (uint32_t)(c * (CHUNK * DMEM * 4)) + (uint32_t)(tid * 16);
#pragma unroll
  for (int i = 0; i < 8; ++i) {
    asm volatile("global_load_async_to_lds_b128 %0, %1, %2"
                 :: "v"(lds0 + (uint32_t)(i * 4096)),
                    "v"(goff + (uint32_t)(i * 4096)),
                    "s"(gbase)
                 : "memory");
  }
}

// ---- stage 0 & 3: OUT = act(A @ W.T + bias), M=128 fixed -----------------
// A: M x K f32 row-major, W: N x K f32 row-major. grid.x = N/64, 256 threads.
__global__ __launch_bounds__(256) void
rm_gemm_xwt(const float* __restrict__ A, const float* __restrict__ W,
            const float* __restrict__ bias, float* __restrict__ OUT,
            int N, int K, int doRelu) {
  __shared__ __align__(16) bf16_t Abf[128][PADK];
  __shared__ __align__(16) bf16_t Wbf[64][PADK];

  const int tid = threadIdx.x;
  const int w = tid >> 5, lane = tid & 31;
  const int n0 = blockIdx.x * 64;

  v8f acc[4] = {{}, {}, {}, {}};

  for (int k0 = 0; k0 < K; k0 += 64) {
#pragma unroll
    for (int i = 0; i < 16; ++i) {               // A tile 128x64, 2 elems/thread
      int idx = (i * 256 + tid) * 2;
      int m = idx >> 6, kx = idx & 63;
      float2 v = *(const float2*)&A[(size_t)m * K + k0 + kx];
      store_pk(&Abf[m][kx], v.x, v.y);
    }
#pragma unroll
    for (int i = 0; i < 8; ++i) {                // W tile 64x64
      int idx = (i * 256 + tid) * 2;
      int nn = idx >> 6, kx = idx & 63;
      float2 v = *(const float2*)&W[(size_t)(n0 + nn) * K + k0 + kx];
      store_pk(&Wbf[nn][kx], v.x, v.y);
    }
    __syncthreads();
#pragma unroll
    for (int kk = 0; kk < 64; kk += 32) {
      v16bf a  = load_frag_a(&Abf[w * 16][0] + kk, PADK, lane);
      v16bf b0 = load_frag_b(&Wbf[0][0] + kk, PADK, lane);
      v16bf b1 = load_frag_b(&Wbf[16][0] + kk, PADK, lane);
      v16bf b2 = load_frag_b(&Wbf[32][0] + kk, PADK, lane);
      v16bf b3 = load_frag_b(&Wbf[48][0] + kk, PADK, lane);
      acc[0] = wmma_bf16(a, b0, acc[0]);
      acc[1] = wmma_bf16(a, b1, acc[1]);
      acc[2] = wmma_bf16(a, b2, acc[2]);
      acc[3] = wmma_bf16(a, b3, acc[3]);
    }
    __syncthreads();
  }

  const int nn = lane & 15;
  const int mo = (lane >> 4) * 8;
#pragma unroll
  for (int nt = 0; nt < 4; ++nt) {
    int ncol = n0 + nt * 16 + nn;
    float bv = bias[ncol];
#pragma unroll
    for (int r = 0; r < 8; ++r) {
      float v = acc[nt][r] + bv;
      if (doRelu) v = v > 0.0f ? v : 0.0f;
      OUT[(size_t)(w * 16 + mo + r) * N + ncol] = v;
    }
  }
}

// ---- stage 1: matrix[b] = relu(mem+addr).T @ mem ; normalizer ------------
// One block per batch, 256 threads (8 waves), async double-buffered pipeline.
__global__ __launch_bounds__(256) void
rm_stage1(const float* __restrict__ memories, const float* __restrict__ addresses,
          float* __restrict__ matrixOut, float* __restrict__ normOut) {
  __shared__ __align__(16) float  stage[2][CHUNK * DMEM];   // 2 x 32KB
  __shared__ __align__(16) bf16_t kT[DMEM][PADC];           // k transposed (d,n)
  __shared__ __align__(16) bf16_t mT[DMEM][PADC];           // mem transposed
  __shared__ float nred[4][DMEM];

  const int b = blockIdx.x;
  const int tid = threadIdx.x;
  const int lane = tid & 31;
  const int w = tid >> 5;
  const float* gbase = memories + (size_t)b * NMEM * DMEM;

  const int dT = w >> 1;          // 0..3 : output row tile
  const int eT0 = (w & 1) * 2;    // 0 or 2: first of two output col tiles

  v8f acc0 = {}, acc1 = {};
  float normAcc = 0.0f;
  const int d = tid & 63;
  const int g = tid >> 6;         // 0..3

  issue_chunk(gbase, 0, &stage[0][0], tid);

  const int NCH = NMEM / CHUNK;   // 32
  for (int c = 0; c < NCH; ++c) {
    const int cur = c & 1;
    if (c + 1 < NCH) {
      issue_chunk(gbase, c + 1, &stage[cur ^ 1][0], tid);
      asm volatile("s_wait_asynccnt 8" ::: "memory");  // current chunk done
    } else {
      asm volatile("s_wait_asynccnt 0" ::: "memory");
    }
    __syncthreads();   // all waves' DMA done; prev WMMA reads of kT/mT done

    const float* abase = addresses + (size_t)c * CHUNK * DMEM;
#pragma unroll
    for (int i = 0; i < 16; ++i) {
      int n = i * 8 + g * 2;            // even n; pair (n, n+1), fixed d
      float m0 = stage[cur][n * 64 + d];
      float m1 = stage[cur][(n + 1) * 64 + d];
      float k0v = m0 + abase[n * 64 + d];
      float k1v = m1 + abase[(n + 1) * 64 + d];
      k0v = k0v > 0.0f ? k0v : 0.0f;
      k1v = k1v > 0.0f ? k1v : 0.0f;
      store_pk(&kT[d][n], k0v, k1v);
      store_pk(&mT[d][n], m0, m1);
      normAcc += k0v + k1v;
    }
    __syncthreads();

#pragma unroll
    for (int kk = 0; kk < CHUNK; kk += 32) {
      v16bf a  = load_frag_a(&kT[dT * 16][0] + kk, PADC, lane);
      v16bf b0 = load_frag_b(&mT[eT0 * 16][0] + kk, PADC, lane);
      v16bf b1 = load_frag_b(&mT[(eT0 + 1) * 16][0] + kk, PADC, lane);
      acc0 = wmma_bf16(a, b0, acc0);
      acc1 = wmma_bf16(a, b1, acc1);
    }
  }

  float* mOut = matrixOut + (size_t)b * DMEM * DMEM;
  const int nn = lane & 15;
  const int mo = (lane >> 4) * 8;
#pragma unroll
  for (int r = 0; r < 8; ++r) {
    mOut[(dT * 16 + mo + r) * DMEM + eT0 * 16 + nn]       = acc0[r];
    mOut[(dT * 16 + mo + r) * DMEM + (eT0 + 1) * 16 + nn] = acc1[r];
  }
  nred[g][d] = normAcc;
  __syncthreads();
  if (tid < DMEM) {
    normOut[(size_t)b * DMEM + tid] =
        nred[0][tid] + nred[1][tid] + nred[2][tid] + nred[3][tid];
  }
}

// ---- stage 2: attn = (relu(q) @ matrix) / (relu(q)·norm + 1e-5) ----------
__global__ __launch_bounds__(256) void
rm_stage2(const float* __restrict__ qa, const float* __restrict__ matrix,
          const float* __restrict__ norm, float* __restrict__ attn) {
  __shared__ float Ms[DMEM][DMEM + 1];
  __shared__ float qs[16][DMEM];
  __shared__ float dens[16];
  const int b = blockIdx.x, tid = threadIdx.x;

#pragma unroll
  for (int i = 0; i < 16; ++i) {
    int idx = i * 256 + tid;
    Ms[idx >> 6][idx & 63] = matrix[(size_t)b * 4096 + idx];
  }
#pragma unroll
  for (int i = 0; i < 4; ++i) {
    int idx = i * 256 + tid;
    qs[idx >> 6][idx & 63] = qa[(size_t)b * 1024 + idx];  // already relu'd
  }
  __syncthreads();
  if (tid < 16) {
    float s = 0.0f;
    for (int dd = 0; dd < DMEM; ++dd) s += qs[tid][dd] * norm[(size_t)b * DMEM + dd];
    dens[tid] = s + 1e-5f;
  }
  __syncthreads();
#pragma unroll
  for (int i = 0; i < 4; ++i) {
    int idx = i * 256 + tid;
    int h = idx >> 6, e = idx & 63;
    float s = 0.0f;
#pragma unroll
    for (int dd = 0; dd < DMEM; ++dd) s += qs[h][dd] * Ms[dd][e];
    attn[(size_t)b * 1024 + idx] = s / dens[h];
  }
}

// ---- launch --------------------------------------------------------------

extern "C" void kernel_launch(void* const* d_in, const int* in_sizes, int n_in,
                              void* d_out, int out_size, void* d_ws, size_t ws_size,
                              hipStream_t stream) {
  (void)in_sizes; (void)n_in; (void)out_size; (void)ws_size;
  const float* query     = (const float*)d_in[0];
  const float* addresses = (const float*)d_in[1];
  const float* memories  = (const float*)d_in[2];
  const float* Wq        = (const float*)d_in[3];
  const float* bq        = (const float*)d_in[4];
  const float* Wm        = (const float*)d_in[5];
  const float* bm        = (const float*)d_in[6];
  float* out = (float*)d_out;

  char* ws = (char*)d_ws;
  float* qa     = (float*)(ws);                               // 128*1024 f32
  float* matrix = (float*)(ws + 512 * 1024);                  // 128*64*64 f32
  float* norm   = (float*)(ws + 512 * 1024 + 2048 * 1024);    // 128*64 f32
  float* attn   = (float*)(ws + 512 * 1024 + 2048 * 1024 + 64 * 1024);

  // qa = relu(query @ Wq.T + bq)   (fused relu: reference applies relu before use)
  rm_gemm_xwt<<<DMODEL / 64, 256, 0, stream>>>(query, Wq, bq, qa, DMODEL, DMODEL, 1);
  // matrix / normalizer (HBM-bound heavy stage)
  rm_stage1<<<B_SZ, 256, 0, stream>>>(memories, addresses, matrix, norm);
  // attention read-out
  rm_stage2<<<B_SZ, 256, 0, stream>>>(qa, matrix, norm, attn);
  // out = attn @ Wm.T + bm
  rm_gemm_xwt<<<DMODEL / 64, 256, 0, stream>>>(attn, Wm, bm, out, DMODEL, DMODEL, 0);
}
